// CausalSelfAttention_66924180407361
// MI455X (gfx1250) — matrix operand order repaired and checked by reading, not compiled, not run on hardware
//
#include <hip/hip_runtime.h>
#include <hip/hip_bf16.h>

typedef __bf16 bf16_t;
typedef __attribute__((ext_vector_type(16))) __bf16 v16bf;
typedef __attribute__((ext_vector_type(8)))  __bf16 v8bf;
typedef __attribute__((ext_vector_type(8)))  float  v8f;

#define MODEL_DIM 1024
#define NUM_HEADS 16
#define HEAD_DIM  64
#define BATCH     4
#define SEQ       2048
#define ATTN_SCALE 0.125f     // 1/sqrt(64)
#define NEG_BIG   (-3.0e38f)

static __device__ __forceinline__ v16bf load_a16(const bf16_t* p) {
  // A-operand: two contiguous 8-element (16B) segments -> one v16bf
  v8bf lo = *(const v8bf*)(p);
  v8bf hi = *(const v8bf*)(p + 8);
  return __builtin_shufflevector(lo, hi, 0,1,2,3,4,5,6,7,8,9,10,11,12,13,14,15);
}

// ---------------------------------------------------------------------------
// Kernel 0a: x (fp32) -> xb (bf16), row-major. float4 reads, 8B writes.
// ---------------------------------------------------------------------------
__global__ __launch_bounds__(256) void convert_x_kernel(
    const float* __restrict__ x, bf16_t* __restrict__ xb)
{
  size_t i = ((size_t)blockIdx.x * 256 + threadIdx.x) * 4;
  float4 v = *(const float4*)(x + i);
  xb[i + 0] = (bf16_t)v.x; xb[i + 1] = (bf16_t)v.y;
  xb[i + 2] = (bf16_t)v.z; xb[i + 3] = (bf16_t)v.w;
}

// ---------------------------------------------------------------------------
// Kernel 0b: w (fp32, [K][N]) -> wT (bf16, [N][K]) via LDS-tiled transpose.
// Makes every WMMA B-operand fetch a contiguous 32B vector load.
// ---------------------------------------------------------------------------
__global__ __launch_bounds__(256) void transpose_convert_kernel(
    const float* __restrict__ w, bf16_t* __restrict__ wT, int K, int N)
{
  __shared__ float tile[32][33];
  int tx = threadIdx.x & 31;
  int ty = threadIdx.x >> 5;              // 0..7
  int ntn = N >> 5;
  int tn = blockIdx.x % ntn;
  int tk = blockIdx.x / ntn;
  int n0 = tn * 32, k0 = tk * 32;
#pragma unroll
  for (int i = 0; i < 4; ++i)
    tile[ty + i * 8][tx] = w[(size_t)(k0 + ty + i * 8) * N + n0 + tx];
  __syncthreads();
#pragma unroll
  for (int i = 0; i < 4; ++i)
    wT[(size_t)(n0 + ty + i * 8) * K + k0 + tx] = (bf16_t)tile[tx][ty + i * 8];
}

// ---------------------------------------------------------------------------
// Kernel 1: QKV GEMM, bf16 WMMA. One wave computes a 16x64 strip (4 N-tiles,
// A reused across 4 WMMAs). All operand fetches are contiguous vector loads.
// Epilogue scatters Q,K head-major and V TRANSPOSED ([B][H][Dh][S]).
// ---------------------------------------------------------------------------
__global__ __launch_bounds__(256) void qkv_gemm_kernel(
    const bf16_t* __restrict__ xb, const bf16_t* __restrict__ wT,
    const float* __restrict__ bias,
    bf16_t* __restrict__ Q, bf16_t* __restrict__ K, bf16_t* __restrict__ VT)
{
  const int NGRP = (3 * MODEL_DIM) / 64;          // 48 strips along N
  int wave = blockIdx.x * 8 + (threadIdx.x >> 5);
  int lane = threadIdx.x & 31;
  int g  = wave % NGRP;
  int tm = wave / NGRP;
  int m0 = tm * 16, n0 = g * 64;
  int n = lane & 15, half = lane >> 4;

  v8f acc[4] = {};
  const bf16_t* arow = xb + (size_t)(m0 + n) * MODEL_DIM;

  for (int kk = 0; kk < MODEL_DIM; kk += 32) {
    v16bf a = load_a16(arow + kk + half * 16);
#pragma unroll
    for (int t = 0; t < 4; ++t) {
      v16bf b = *(const v16bf*)(wT + (size_t)(n0 + t * 16 + n) * MODEL_DIM
                                   + kk + 16 * half);
      acc[t] = __builtin_amdgcn_wmma_f32_16x16x32_bf16(false, a, false, b,
                                                       (short)0, acc[t], false, false);
    }
  }

#pragma unroll
  for (int t = 0; t < 4; ++t) {
    int ncol = n0 + t * 16 + n;
    float bv = bias[ncol];
    int which = ncol >> 10;              // strip of 64 never straddles Q/K/V
    int rem   = ncol & 1023;
    int h     = rem >> 6;
    int dh    = rem & 63;
#pragma unroll
    for (int r = 0; r < 8; ++r) {
      int mrow = m0 + r + 8 * half;
      int bb = mrow >> 11, s = mrow & (SEQ - 1);
      float v = acc[t][r] + bv;
      if (which == 0)
        Q[(((size_t)bb * NUM_HEADS + h) * SEQ + s) * HEAD_DIM + dh] = (bf16_t)v;
      else if (which == 1)
        K[(((size_t)bb * NUM_HEADS + h) * SEQ + s) * HEAD_DIM + dh] = (bf16_t)v;
      else
        VT[(((size_t)bb * NUM_HEADS + h) * HEAD_DIM + dh) * SEQ + s] = (bf16_t)v;
    }
  }
}

// ---------------------------------------------------------------------------
// Kernel 2: in-place RoPE on Q and K. One thread per rotation pair.
// ---------------------------------------------------------------------------
__global__ __launch_bounds__(256) void rope_kernel(bf16_t* __restrict__ Q,
                                                   bf16_t* __restrict__ K)
{
  const size_t halfTotal = (size_t)BATCH * NUM_HEADS * SEQ * 32;
  size_t idx = (size_t)blockIdx.x * blockDim.x + threadIdx.x;
  if (idx >= 2 * halfTotal) return;
  bf16_t* buf = (idx < halfTotal) ? Q : K;
  size_t id = (idx < halfTotal) ? idx : idx - halfTotal;
  int i = (int)(id & 31);
  size_t rest = id >> 5;
  int s = (int)(rest % SEQ);
  size_t bh = rest / SEQ;
  bf16_t* p = buf + (bh * SEQ + s) * HEAD_DIM;
  float freq = __powf(10000.0f, -(float)i / 32.0f);
  float ang = (float)s * freq;
  float c = __cosf(ang), sn = __sinf(ang);
  float x1 = (float)p[i], x2 = (float)p[i + 32];
  p[i]      = (bf16_t)(x1 * c - x2 * sn);
  p[i + 32] = (bf16_t)(x2 * c + x1 * sn);
}

// ---------------------------------------------------------------------------
// Kernel 3: causal attention. One wave owns 16 query rows of one (b,h).
// Pass A: streaming softmax stats. Pass B: recompute scores, write full attn
// (zeros above diagonal), accumulate P@V with WMMA (V read from VT, all
// contiguous 32B loads).
// ---------------------------------------------------------------------------
__global__ __launch_bounds__(256) void attention_kernel(
    const bf16_t* __restrict__ Q, const bf16_t* __restrict__ K,
    const bf16_t* __restrict__ VT, float* __restrict__ attn,
    bf16_t* __restrict__ AO)
{
  __shared__ float pstage[8][16][32];     // per-wave P transpose staging (16KB)
  int wid = threadIdx.x >> 5, lane = threadIdx.x & 31;
  int wave = blockIdx.x * 8 + wid;
  int qt = wave & 127;
  int bh = wave >> 7;
  int q0 = qt * 16;
  int n = lane & 15, half = lane >> 4;

  const bf16_t* Qb  = Q  + (size_t)bh * SEQ * HEAD_DIM;
  const bf16_t* Kb  = K  + (size_t)bh * SEQ * HEAD_DIM;
  const bf16_t* VTb = VT + (size_t)bh * HEAD_DIM * SEQ;

  v16bf a0, a1;
  {
    const bf16_t* qrow = Qb + (size_t)(q0 + n) * HEAD_DIM;
    a0 = load_a16(qrow + half * 16);
    a1 = load_a16(qrow + 32 + half * 16);
  }

  float mrow[8], lrow[8];
#pragma unroll
  for (int r = 0; r < 8; ++r) { mrow[r] = NEG_BIG; lrow[r] = 0.0f; }

  int kc_end = (q0 + 15) >> 5;

  // ---------------- Pass A: softmax statistics -----------------------------
  for (int kc = 0; kc <= kc_end; ++kc) {
    int kv0 = kc * 32;
    __builtin_prefetch(Kb + (size_t)(kv0 + 32 + n) * HEAD_DIM, 0, 1);
#pragma unroll
    for (int sub = 0; sub < 2; ++sub) {
      int kvs = kv0 + sub * 16;
      const bf16_t* krow = Kb + (size_t)(kvs + n) * HEAD_DIM + 16 * half;
      v16bf b0 = *(const v16bf*)(krow);
      v16bf b1 = *(const v16bf*)(krow + 32);
      v8f c = {};
      c = __builtin_amdgcn_wmma_f32_16x16x32_bf16(false, a0, false, b0, (short)0, c, false, false);
      c = __builtin_amdgcn_wmma_f32_16x16x32_bf16(false, a1, false, b1, (short)0, c, false, false);
#pragma unroll
      for (int r = 0; r < 8; ++r) {
        int qr = q0 + r + 8 * half;
        float s = ((kvs + n) > qr) ? NEG_BIG : c[r] * ATTN_SCALE;
        float mx = s;
#pragma unroll
        for (int off = 1; off < 16; off <<= 1)
          mx = fmaxf(mx, __shfl_xor(mx, off, 32));
        float mnew = fmaxf(mrow[r], mx);
        float e = __expf(s - mnew);
#pragma unroll
        for (int off = 1; off < 16; off <<= 1)
          e += __shfl_xor(e, off, 32);
        lrow[r] = lrow[r] * __expf(mrow[r] - mnew) + e;
        mrow[r] = mnew;
      }
    }
  }

  // ---------------- Pass B: write attn, accumulate P@V ---------------------
  v8f o[4] = {};
  for (int kc = 0; kc < SEQ / 32; ++kc) {
    int kv0 = kc * 32;
    if (kv0 > q0 + 15) {                  // fully above diagonal: attn = 0
#pragma unroll
      for (int sub = 0; sub < 2; ++sub)
#pragma unroll
        for (int r = 0; r < 8; ++r) {
          int qr = q0 + r + 8 * half;
          attn[((size_t)bh * SEQ + qr) * SEQ + kv0 + sub * 16 + n] = 0.0f;
        }
      continue;
    }
#pragma unroll
    for (int sub = 0; sub < 2; ++sub) {
      int kvs = kv0 + sub * 16;
      const bf16_t* krow = Kb + (size_t)(kvs + n) * HEAD_DIM + 16 * half;
      v16bf b0 = *(const v16bf*)(krow);
      v16bf b1 = *(const v16bf*)(krow + 32);
      v8f c = {};
      c = __builtin_amdgcn_wmma_f32_16x16x32_bf16(false, a0, false, b0, (short)0, c, false, false);
      c = __builtin_amdgcn_wmma_f32_16x16x32_bf16(false, a1, false, b1, (short)0, c, false, false);
#pragma unroll
      for (int r = 0; r < 8; ++r) {
        int qr = q0 + r + 8 * half;
        float p = 0.0f;
        if ((kvs + n) <= qr)
          p = __expf(c[r] * ATTN_SCALE - mrow[r]) / lrow[r];
        attn[((size_t)bh * SEQ + qr) * SEQ + kvs + n] = p;
        pstage[wid][r + 8 * half][sub * 16 + n] = p;
      }
    }
    // wave-private LDS transpose: D-layout (col-per-lane) -> A-layout (row-per-lane)
    asm volatile("s_wait_dscnt 0" ::: "memory");
    v16bf pa;
    {
      int base = half * 16;
#pragma unroll
      for (int j = 0; j < 8; ++j) {
        pa[j]     = (bf16_t)pstage[wid][n][base + j];
        pa[j + 8] = (bf16_t)pstage[wid][n][base + 8 + j];
      }
    }
#pragma unroll
    for (int d = 0; d < 4; ++d) {
      // V B-operand from VT: contiguous 32B along kv
      v16bf vb = *(const v16bf*)(VTb + (size_t)(d * 16 + n) * SEQ + kv0 + 16 * half);
      o[d] = __builtin_amdgcn_wmma_f32_16x16x32_bf16(false, pa, false, vb,
                                                     (short)0, o[d], false, false);
    }
  }

  int b = bh >> 4, h = bh & 15;
#pragma unroll
  for (int d = 0; d < 4; ++d)
#pragma unroll
    for (int r = 0; r < 8; ++r) {
      int s = q0 + r + 8 * half;
      AO[((size_t)b * SEQ + s) * MODEL_DIM + h * HEAD_DIM + d * 16 + n] =
          (bf16_t)o[d][r];
    }
}

// ---------------------------------------------------------------------------
// Kernel 4: out = AO @ w_out + b_out (fp32). 16x64 strip per wave.
// ---------------------------------------------------------------------------
__global__ __launch_bounds__(256) void out_proj_kernel(
    const bf16_t* __restrict__ A, const bf16_t* __restrict__ wT,
    const float* __restrict__ bias, float* __restrict__ out)
{
  const int NGRP = MODEL_DIM / 64;        // 16 strips
  int wave = blockIdx.x * 8 + (threadIdx.x >> 5);
  int lane = threadIdx.x & 31;
  int g  = wave % NGRP;
  int tm = wave / NGRP;
  int m0 = tm * 16, n0 = g * 64;
  int n = lane & 15, half = lane >> 4;

  v8f acc[4] = {};
  const bf16_t* arow = A + (size_t)(m0 + n) * MODEL_DIM;
  for (int kk = 0; kk < MODEL_DIM; kk += 32) {
    v16bf a = load_a16(arow + kk + half * 16);
#pragma unroll
    for (int t = 0; t < 4; ++t) {
      v16bf b = *(const v16bf*)(wT + (size_t)(n0 + t * 16 + n) * MODEL_DIM
                                   + kk + 16 * half);
      acc[t] = __builtin_amdgcn_wmma_f32_16x16x32_bf16(false, a, false, b,
                                                       (short)0, acc[t], false, false);
    }
  }
#pragma unroll
  for (int t = 0; t < 4; ++t) {
    int ncol = n0 + t * 16 + n;
    float bv = bias[ncol];
#pragma unroll
    for (int r = 0; r < 8; ++r)
      out[(size_t)(m0 + r + 8 * half) * MODEL_DIM + ncol] = acc[t][r] + bv;
  }
}

// ---------------------------------------------------------------------------
extern "C" void kernel_launch(void* const* d_in, const int* in_sizes, int n_in,
                              void* d_out, int out_size, void* d_ws, size_t ws_size,
                              hipStream_t stream) {
  const float* x     = (const float*)d_in[0];
  const float* w_qkv = (const float*)d_in[1];
  const float* b_qkv = (const float*)d_in[2];
  const float* w_out = (const float*)d_in[3];
  const float* b_out = (const float*)d_in[4];

  float* out  = (float*)d_out;
  float* attn = out + (size_t)BATCH * SEQ * MODEL_DIM;

  const size_t M = (size_t)BATCH * SEQ;                 // 8192
  const size_t qkv_elems = (size_t)BATCH * NUM_HEADS * SEQ * HEAD_DIM;
  bf16_t* xb     = (bf16_t*)d_ws;                       // [M][Dm]
  bf16_t* wqkvT  = xb + M * MODEL_DIM;                  // [3Dm][Dm]
  bf16_t* woutT  = wqkvT + (size_t)3 * MODEL_DIM * MODEL_DIM;  // [Dm][Dm]
  bf16_t* Q      = woutT + (size_t)MODEL_DIM * MODEL_DIM;
  bf16_t* K      = Q + qkv_elems;
  bf16_t* VT     = K + qkv_elems;                       // [B][H][Dh][S]
  bf16_t* AO     = VT + qkv_elems;                      // [B][S][Dm]

  // 0) precision/layout staging
  convert_x_kernel<<<(unsigned)(M * MODEL_DIM / 1024), 256, 0, stream>>>(x, xb);
  transpose_convert_kernel<<<(1024 / 32) * (3 * MODEL_DIM / 32), 256, 0, stream>>>(
      w_qkv, wqkvT, MODEL_DIM, 3 * MODEL_DIM);
  transpose_convert_kernel<<<(1024 / 32) * (MODEL_DIM / 32), 256, 0, stream>>>(
      w_out, woutT, MODEL_DIM, MODEL_DIM);

  // 1) QKV projection: 512 m-tiles * 48 strips = 24576 waves -> 3072 blocks
  qkv_gemm_kernel<<<3072, 256, 0, stream>>>(xb, wqkvT, b_qkv, Q, K, VT);

  // 2) RoPE
  rope_kernel<<<32768, 256, 0, stream>>>(Q, K);

  // 3) Attention: 8192 waves -> 1024 blocks
  attention_kernel<<<1024, 256, 0, stream>>>(Q, K, VT, attn, AO);

  // 4) Output projection: 512 * 16 = 8192 waves -> 1024 blocks
  out_proj_kernel<<<1024, 256, 0, stream>>>(AO, woutT, b_out, out);
}
